// MultiheadAttention_10264971837452
// MI455X (gfx1250) — matrix operand order repaired and checked by reading, not compiled, not run on hardware
//
#include <hip/hip_runtime.h>
#include <hip/hip_bf16.h>

// Problem constants (from reference)
#define NB   2
#define TT   2048
#define SSZ  2048
#define EE   1024
#define HH   16
#define PPD  16
#define EHD  64
#define SCALE_F 0.11180339887498948f   // 1/sqrt(64+16)

typedef __bf16 bf16_t;
typedef __attribute__((ext_vector_type(16))) __bf16 v16bf;
typedef __attribute__((ext_vector_type(8)))  float  v8f;
typedef __attribute__((ext_vector_type(4)))  int    v4i;

union BF16Frag { v16bf v; uint4 u[2]; };
union F8       { v8f v; float f[8]; };

#if defined(__gfx1250__) && __has_builtin(__builtin_amdgcn_global_load_async_to_lds_b128)
#define HAVE_ASYNC_LDS 1
typedef __attribute__((address_space(1))) v4i* gp128_t;   // global int4*
typedef __attribute__((address_space(3))) v4i* lp128_t;   // LDS int4*
#else
#define HAVE_ASYNC_LDS 0
#endif

__device__ __forceinline__ v8f wmma_bf16(v16bf a, v16bf b, v8f c) {
  return __builtin_amdgcn_wmma_f32_16x16x32_bf16(false, a, false, b, (short)0, c, false, false);
}

__device__ __forceinline__ unsigned pkbf(float a, float b) {
  union { bf16_t h[2]; unsigned u; } x;
  x.h[0] = (bf16_t)a; x.h[1] = (bf16_t)b;
  return x.u;
}

__device__ __forceinline__ v16bf cvt16(float4 a, float4 b, float4 c, float4 d) {
  v16bf v;
  v[0]=(bf16_t)a.x;  v[1]=(bf16_t)a.y;  v[2]=(bf16_t)a.z;  v[3]=(bf16_t)a.w;
  v[4]=(bf16_t)b.x;  v[5]=(bf16_t)b.y;  v[6]=(bf16_t)b.z;  v[7]=(bf16_t)b.w;
  v[8]=(bf16_t)c.x;  v[9]=(bf16_t)c.y;  v[10]=(bf16_t)c.z; v[11]=(bf16_t)c.w;
  v[12]=(bf16_t)d.x; v[13]=(bf16_t)d.y; v[14]=(bf16_t)d.z; v[15]=(bf16_t)d.w;
  return v;
}

__device__ __forceinline__ float dot16(const float4* pq, const float* p) {
  const float4* q = (const float4*)p;
  float4 a = q[0], b = q[1], c = q[2], d = q[3];
  return pq[0].x*a.x + pq[0].y*a.y + pq[0].z*a.z + pq[0].w*a.w
       + pq[1].x*b.x + pq[1].y*b.y + pq[1].z*b.z + pq[1].w*b.w
       + pq[2].x*c.x + pq[2].y*c.y + pq[2].z*c.z + pq[2].w*c.w
       + pq[3].x*d.x + pq[3].y*d.y + pq[3].z*d.z + pq[3].w*d.w;
}

// ---------------------------------------------------------------------------
// Generic GEMM: C = A @ W^T.   A: M x K (f32 or bf16, row-major),
// W: N x K f32 row-major (converted to bf16 on the fly).
// One wave computes a 16x32 tile (two N-subtiles share one A fragment).
// OUTMODE 0: bf16 row-major [M][N]
// OUTMODE 1: bf16 transposed per batch: out[nb][col][s], nb = m/Sdim, s = m%Sdim
// OUTMODE 2: f32 row-major [M][N]
// 4 waves (128 threads) per block stacked in M.
// ---------------------------------------------------------------------------
template<int OUTMODE, bool ABF16>
__global__ __launch_bounds__(128) void gemm_xwt(const void* __restrict__ Aptr,
                                                const float* __restrict__ W,
                                                void* __restrict__ outp,
                                                int M, int N, int K, int Sdim) {
  const int lane  = threadIdx.x & 31;
  const int wid   = threadIdx.x >> 5;
  const int m0    = (blockIdx.y * 4 + wid) * 16;
  const int n0    = blockIdx.x * 32;
  const int half  = lane >> 4;     // 0 | 1
  const int l16   = lane & 15;
  const int base8 = half * 8;
  const int kbA   = half * 16;     // A-frag K sub-offset (contiguous K, matches B)
  const int kbB   = half * 16;     // B-frag K sub-offset (contiguous K, matches A)

  const int arow = m0 + l16;

  v8f acc0 = {0.f,0.f,0.f,0.f,0.f,0.f,0.f,0.f};
  v8f acc1 = {0.f,0.f,0.f,0.f,0.f,0.f,0.f,0.f};

  for (int k0 = 0; k0 < K; k0 += 32) {
    BF16Frag a, b0, b1;
    if (ABF16) {
      const bf16_t* A = (const bf16_t*)Aptr;
      const bf16_t* ap = A + (size_t)arow * K + k0 + kbA;
      a.u[0] = *(const uint4*)(ap);
      a.u[1] = *(const uint4*)(ap + 8);
    } else {
      const float* A = (const float*)Aptr;
      const float* ap = A + (size_t)arow * K + k0 + kbA;
      float4 a0 = ((const float4*)ap)[0];
      float4 a1 = ((const float4*)ap)[1];
      float4 a2 = ((const float4*)ap)[2];
      float4 a3 = ((const float4*)ap)[3];
      a.v = cvt16(a0, a1, a2, a3);
    }
    {
      const float* wp = W + (size_t)(n0 + l16) * K + k0 + kbB;
      float4 w0 = ((const float4*)wp)[0];
      float4 w1 = ((const float4*)wp)[1];
      float4 w2 = ((const float4*)wp)[2];
      float4 w3 = ((const float4*)wp)[3];
      b0.v = cvt16(w0, w1, w2, w3);
    }
    {
      const float* wp = W + (size_t)(n0 + 16 + l16) * K + k0 + kbB;
      float4 w0 = ((const float4*)wp)[0];
      float4 w1 = ((const float4*)wp)[1];
      float4 w2 = ((const float4*)wp)[2];
      float4 w3 = ((const float4*)wp)[3];
      b1.v = cvt16(w0, w1, w2, w3);
    }
    acc0 = wmma_bf16(a.v, b0.v, acc0);
    acc1 = wmma_bf16(a.v, b1.v, acc1);
  }

#pragma unroll
  for (int j = 0; j < 2; ++j) {
    F8 c; c.v = (j == 0) ? acc0 : acc1;
    const int col = n0 + j * 16 + l16;
    if (OUTMODE == 0) {
      bf16_t* O = (bf16_t*)outp;
#pragma unroll
      for (int r = 0; r < 8; ++r)
        O[(size_t)(m0 + base8 + r) * N + col] = (bf16_t)c.f[r];
    } else if (OUTMODE == 1) {
      bf16_t* O = (bf16_t*)outp;
      const int nb = m0 / Sdim;
      const int sb = (m0 % Sdim) + base8;
      uint4 u;
      u.x = pkbf(c.f[0], c.f[1]);
      u.y = pkbf(c.f[2], c.f[3]);
      u.z = pkbf(c.f[4], c.f[5]);
      u.w = pkbf(c.f[6], c.f[7]);
      *(uint4*)(O + (size_t)nb * N * Sdim + (size_t)col * Sdim + sb) = u;
    } else {
      float* O = (float*)outp;
#pragma unroll
      for (int r = 0; r < 8; ++r)
        O[(size_t)(m0 + base8 + r) * N + col] = c.f[r];
    }
  }
}

// ---------------------------------------------------------------------------
// Flash attention with additive prior.
// Grid: (TT/16, NB). Block: 512 threads = 16 waves; wave w handles head h=w.
// Computes L^T = K·Q^T (lane = query column t) so softmax stats need only one
// shfl_xor(16); exp'd tiles reassemble directly into the B-fragment of
// O^T = V^T · P^T with 8 shuffles. Prior (t-dependent, non-GEMM) via LDS +
// per-lane dot16; prior/mask tiles shared by all 16 heads (read once).
// Prior staging uses GLOBAL_LOAD_ASYNC_TO_LDS (ASYNCcnt + s_wait_asynccnt).
// ---------------------------------------------------------------------------
#define PRS 516   // padded float stride per t-row of prior tile (32 s * 16 p + 4)
#define MSK 36    // padded float stride per t-row of mask tile

__global__ __launch_bounds__(512) void attn_kernel(const bf16_t* __restrict__ Q,
                                                   const bf16_t* __restrict__ Kc,
                                                   const bf16_t* __restrict__ Vt,
                                                   const float*  __restrict__ Pq,
                                                   const float*  __restrict__ prior,
                                                   const float*  __restrict__ mask,
                                                   bf16_t* __restrict__ O) {
  __shared__ float prLds[16 * PRS];
  __shared__ float mkLds[16 * MSK];

  const int lane  = threadIdx.x & 31;
  const int h     = threadIdx.x >> 5;
  const int t0    = blockIdx.x * 16;
  const int nb    = blockIdx.y;
  const int half  = lane >> 4;
  const int l16   = lane & 15;
  const int base8 = half * 8;
  const int t     = t0 + l16;      // this lane's query column

  // Q fragments (B operand of L^T), two K-steps over EH=64
  BF16Frag qf0, qf1;
  {
    const bf16_t* qp = Q + (size_t)(nb * TT + t) * EE + h * EHD + half * 16;
    qf0.u[0] = *(const uint4*)(qp);
    qf0.u[1] = *(const uint4*)(qp + 8);
    qf1.u[0] = *(const uint4*)(qp + 32);
    qf1.u[1] = *(const uint4*)(qp + 40);
  }

  // prior-query vector for (n,h,t,:) with the faithful reshape quirk:
  // prior_q[n,h,t,p] = PqProj[row = 2*t + n][h*16 + p]
  float4 pq4[4];
  {
    const float4* pqp = (const float4*)(Pq + (size_t)(2 * t + nb) * (HH * PPD) + h * PPD);
    pq4[0] = pqp[0]; pq4[1] = pqp[1]; pq4[2] = pqp[2]; pq4[3] = pqp[3];
  }

  float m_run = -__builtin_inff();
  float l_run = 0.f;
  F8 acc[4];
#pragma unroll
  for (int e = 0; e < 4; ++e)
#pragma unroll
    for (int r = 0; r < 8; ++r) acc[e].f[r] = 0.f;

  for (int s0 = 0; s0 < SSZ; s0 += 32) {
    // ---- stage prior + mask tiles (all 512 threads; shared by all heads) ----
    {
      const int tl = h;            // wave id 0..15 = t row
      const int sl = lane;         // 0..31 = s column
      const float* gp = prior + ((size_t)(nb * TT + t0 + tl) * SSZ + (s0 + sl)) * PPD;
      const float* gm = mask + (size_t)(nb * TT + t0 + tl) * SSZ + s0 + sl;
      float* dst = &prLds[tl * PRS + sl * 16];
#if HAVE_ASYNC_LDS
      {
        gp128_t g = (gp128_t)gp;
        lp128_t l = (lp128_t)dst;
        __builtin_amdgcn_global_load_async_to_lds_b128(g, l, 0, 0);
        __builtin_amdgcn_global_load_async_to_lds_b128(g, l, 16, 0);
        __builtin_amdgcn_global_load_async_to_lds_b128(g, l, 32, 0);
        __builtin_amdgcn_global_load_async_to_lds_b128(g, l, 48, 0);
        mkLds[tl * MSK + sl] = *gm;
        asm volatile("s_wait_asynccnt 0" ::: "memory");
      }
#else
      {
        float4 p0 = ((const float4*)gp)[0];
        float4 p1 = ((const float4*)gp)[1];
        float4 p2 = ((const float4*)gp)[2];
        float4 p3 = ((const float4*)gp)[3];
        float4* d4 = (float4*)dst;
        d4[0] = p0; d4[1] = p1; d4[2] = p2; d4[3] = p3;
        mkLds[tl * MSK + sl] = *gm;
      }
#endif
      if (s0 + 32 < SSZ) __builtin_prefetch(gp + 32 * PPD, 0, 1);
    }
    __syncthreads();

    // ---- content logits L^T = K · Q^T : two 16x16 C tiles (s-rows) ----
    const bf16_t* kp0 = Kc + (size_t)(nb * SSZ + s0 + l16) * EE + h * EHD;
    const bf16_t* kp1 = kp0 + (size_t)16 * EE;
    BF16Frag kf00, kf01, kf10, kf11;
    kf00.u[0] = *(const uint4*)(kp0 + half * 16);
    kf00.u[1] = *(const uint4*)(kp0 + 8 + half * 16);
    kf01.u[0] = *(const uint4*)(kp0 + 32 + half * 16);
    kf01.u[1] = *(const uint4*)(kp0 + 40 + half * 16);
    kf10.u[0] = *(const uint4*)(kp1 + half * 16);
    kf10.u[1] = *(const uint4*)(kp1 + 8 + half * 16);
    kf11.u[0] = *(const uint4*)(kp1 + 32 + half * 16);
    kf11.u[1] = *(const uint4*)(kp1 + 40 + half * 16);

    v8f z = {0.f,0.f,0.f,0.f,0.f,0.f,0.f,0.f};
    F8 c0, c1;
    c0.v = wmma_bf16(kf00.v, qf0.v, z);
    c0.v = wmma_bf16(kf01.v, qf1.v, c0.v);
    c1.v = wmma_bf16(kf10.v, qf0.v, z);
    c1.v = wmma_bf16(kf11.v, qf1.v, c1.v);

    // ---- add prior + mask, scale ----
#pragma unroll
    for (int r = 0; r < 8; ++r) {
      const int sA = base8 + r;
      const int sB = 16 + base8 + r;
      float dA = dot16(pq4, &prLds[l16 * PRS + sA * 16]);
      float dB = dot16(pq4, &prLds[l16 * PRS + sB * 16]);
      c0.f[r] = (c0.f[r] + dA) * SCALE_F + mkLds[l16 * MSK + sA];
      c1.f[r] = (c1.f[r] + dB) * SCALE_F + mkLds[l16 * MSK + sB];
    }

    // ---- online softmax per query column t ----
    float mx = c0.f[0];
#pragma unroll
    for (int r = 1; r < 8; ++r) mx = fmaxf(mx, c0.f[r]);
#pragma unroll
    for (int r = 0; r < 8; ++r) mx = fmaxf(mx, c1.f[r]);
    mx = fmaxf(mx, __shfl_xor(mx, 16));
    const float m_new = fmaxf(m_run, mx);
    const float corr  = __expf(m_run - m_new);
    float ls = 0.f;
#pragma unroll
    for (int r = 0; r < 8; ++r) { c0.f[r] = __expf(c0.f[r] - m_new); ls += c0.f[r]; }
#pragma unroll
    for (int r = 0; r < 8; ++r) { c1.f[r] = __expf(c1.f[r] - m_new); ls += c1.f[r]; }
    ls += __shfl_xor(ls, 16);
    l_run = l_run * corr + ls;
    m_run = m_new;
#pragma unroll
    for (int e = 0; e < 4; ++e)
#pragma unroll
      for (int r = 0; r < 8; ++r) acc[e].f[r] *= corr;

    // ---- assemble P^T B-fragment (lanes 0-15: k=s 0..15, lanes 16-31: 16..31)
    float recv[8];
#pragma unroll
    for (int r = 0; r < 8; ++r) {
      float snd = half ? c0.f[r] : c1.f[r];
      recv[r] = __shfl_xor(snd, 16);
    }
    v16bf pf;
#pragma unroll
    for (int j = 0; j < 8; ++j) {
      float lo = half ? recv[j]  : c0.f[j];
      float hi = half ? c1.f[j]  : recv[j];
      pf[j]     = (bf16_t)lo;
      pf[8 + j] = (bf16_t)hi;
    }

    // ---- O^T += V^T · P^T : 4 eh-subtiles ----
#pragma unroll
    for (int e = 0; e < 4; ++e) {
      const bf16_t* vp = Vt + ((size_t)nb * EE + h * EHD + e * 16 + l16) * SSZ + s0;
      BF16Frag vf;
      vf.u[0] = *(const uint4*)(vp + half * 16);
      vf.u[1] = *(const uint4*)(vp + 8 + half * 16);
      acc[e].v = wmma_bf16(vf.v, pf, acc[e].v);
    }
    __syncthreads();
  }

  // ---- normalize and store O (bf16 row-major (n,t,h,eh)) ----
  const float inv = 1.0f / l_run;
  bf16_t* op = O + (size_t)(nb * TT + t) * EE + h * EHD;
#pragma unroll
  for (int e = 0; e < 4; ++e) {
    uint4 u;
    u.x = pkbf(acc[e].f[0] * inv, acc[e].f[1] * inv);
    u.y = pkbf(acc[e].f[2] * inv, acc[e].f[3] * inv);
    u.z = pkbf(acc[e].f[4] * inv, acc[e].f[5] * inv);
    u.w = pkbf(acc[e].f[6] * inv, acc[e].f[7] * inv);
    *(uint4*)(op + e * 16 + base8) = u;
  }
}

// ---------------------------------------------------------------------------
extern "C" void kernel_launch(void* const* d_in, const int* in_sizes, int n_in,
                              void* d_out, int out_size, void* d_ws, size_t ws_size,
                              hipStream_t stream) {
  (void)in_sizes; (void)n_in; (void)out_size; (void)ws_size;
  const float* Xq    = (const float*)d_in[0];
  const float* Xk    = (const float*)d_in[1];
  const float* prior = (const float*)d_in[2];
  const float* mask  = (const float*)d_in[3];
  const float* Wq    = (const float*)d_in[4];
  const float* Wk    = (const float*)d_in[5];
  const float* Wv    = (const float*)d_in[6];
  const float* Wp    = (const float*)d_in[7];
  const float* Wo    = (const float*)d_in[8];

  char* ws = (char*)d_ws;
  bf16_t* Qb  = (bf16_t*)(ws);                       // 8 MB
  bf16_t* Kb  = (bf16_t*)(ws + ((size_t)8  << 20));  // 8 MB
  bf16_t* Vtb = (bf16_t*)(ws + ((size_t)16 << 20));  // 8 MB
  float*  Pqb = (float*) (ws + ((size_t)24 << 20));  // 4 MB
  bf16_t* Ob  = (bf16_t*)(ws + ((size_t)28 << 20));  // 8 MB

  const int M = NB * TT;                 // 4096 rows
  dim3 blk(128);
  dim3 gE(EE / 32, M / 64);              // 32 x 64
  dim3 gP((HH * PPD) / 32, M / 64);      // 8 x 64

  gemm_xwt<0, false><<<gE, blk, 0, stream>>>(Xq, Wq, Qb,  M, EE,       EE, SSZ);
  gemm_xwt<0, false><<<gE, blk, 0, stream>>>(Xk, Wk, Kb,  M, EE,       EE, SSZ);
  gemm_xwt<1, false><<<gE, blk, 0, stream>>>(Xk, Wv, Vtb, M, EE,       EE, SSZ);
  gemm_xwt<2, false><<<gP, blk, 0, stream>>>(Xq, Wp, Pqb, M, HH * PPD, EE, SSZ);

  attn_kernel<<<dim3(TT / 16, NB), 512, 0, stream>>>(Qb, Kb, Vtb, Pqb, prior, mask, Ob);

  gemm_xwt<2, true><<<gE, blk, 0, stream>>>(Ob, Wo, d_out, M, EE, EE, SSZ);
}
